// DFlashModel_4776003633810
// MI455X (gfx1250) — matrix operand order repaired and checked by reading, not compiled, not run on hardware
//
#include <hip/hip_runtime.h>
#include <hip/hip_bf16.h>

// ---------------- problem constants (from reference) ----------------
#define BB     2
#define SS     2048
#define DD     2048
#define LL     3
#define HH     16
#define DHH    128          // D / H
#define BLK    16
#define NAA    64
#define QQ     1024         // NA * BLOCK
#define KVV    3072         // S + Q
#define VOC    32000
#define MASKID 31999
#define GAMMA_ 7.0f
#define EPS_   1e-6f
#define CHUNKW 2048         // lm-head chunk columns (multiple of 128)
#define NCHUNK 16           // 15*2048 + 1*1280 = 32000

typedef __attribute__((ext_vector_type(16))) _Float16 v16h;
typedef __attribute__((ext_vector_type(8)))  float    v8f;
typedef __attribute__((ext_vector_type(4)))  int      v4i;

union HV { v16h v; _Float16 h[16]; unsigned int u[8]; uint4 q[2]; };

// ---------------- async global->LDS (CDNA5) with sync fallback -------
#if defined(__has_builtin)
#if __has_builtin(__builtin_amdgcn_global_load_async_to_lds_b128)
#define USE_ASYNC 1
#endif
#endif
#ifndef USE_ASYNC
#define USE_ASYNC 0
#endif

#if USE_ASYNC
typedef __attribute__((address_space(1))) v4i* gv4p;   // global int4*
typedef __attribute__((address_space(3))) v4i* lv4p;   // LDS int4*
__device__ __forceinline__ void async_cp16(const _Float16* g, _Float16* l)
{
    __builtin_amdgcn_global_load_async_to_lds_b128((gv4p)g, (lv4p)l, 0, 0);
}
__device__ __forceinline__ void async_wait0()
{
#if __has_builtin(__builtin_amdgcn_s_wait_asynccnt)
    __builtin_amdgcn_s_wait_asynccnt(0);
#else
    asm volatile("s_wait_asynccnt 0x0" ::: "memory");
#endif
}
#else
__device__ __forceinline__ void async_cp16(const _Float16*, _Float16*) {}
__device__ __forceinline__ void async_wait0() {}
#endif

// =====================================================================
// Blocked WMMA GEMM:  C[M,N] = A[M,K] * B'   (f16 in, f32 out)
//   TRANSB=1:  B given as W[N,K] row-major (x @ W^T)
//   TRANSB=0:  B given as [K,N]  row-major
// Block tile 64(M) x 128(N), 8 waves in 2x4 grid, each wave owns a
// 32x32 sub-tile = 2x2 WMMA tiles.  K staged through double-buffered
// LDS in 32-wide steps; async global->LDS path when available.
// Requires M%64==0, N%128==0, K%32==0 (true at every call site).
// =====================================================================
template<int TRANSB>
__global__ __launch_bounds__(256)
void gemm_wmma_blk(const _Float16* __restrict__ A, long long lda, long long sAz,
                   const _Float16* __restrict__ Bm, long long ldb, long long sBz,
                   float* __restrict__ C, long long ldc, long long sCz,
                   int M, int N, int K)
{
    __shared__ __align__(16) _Float16 Ash[2][64 * 32];
    __shared__ __align__(16) _Float16 Bsh[2][128 * 32];

    const int t     = threadIdx.x;
    const int lane  = t & 31;
    const int wave  = t >> 5;
    const int wm    = wave >> 2;         // 0..1  (M slab)
    const int wn    = wave & 3;          // 0..3  (N slab)
    const int r16   = lane & 15;
    const int khalf = lane >> 4;

    const int nblk = N >> 7;
    const int m0 = (blockIdx.x / nblk) * 64;
    const int n0 = (blockIdx.x % nblk) * 128;

    A  += (long long)blockIdx.z * sAz;
    Bm += (long long)blockIdx.z * sBz;
    C  += (long long)blockIdx.z * sCz;

    // staging decomposition: thread -> (row 0..63, 16-byte chunk 0..3)
    const int arow = t >> 2;
    const int achk = (t & 3) << 3;       // element offset of 8-f16 chunk

    const _Float16* Ag  = A  + (long long)(m0 + arow) * lda + achk;
    const _Float16* Bg0 = Bm + (long long)(n0 + arow) * ldb + achk;        // TRANSB=1
    const _Float16* Bg1 = Bm + (long long)(n0 + arow + 64) * ldb + achk;   // TRANSB=1

    _Float16* aSlot[2] = { &Ash[0][arow * 32 + achk], &Ash[1][arow * 32 + achk] };
    _Float16* bSlot0[2] = { &Bsh[0][arow * 32 + achk], &Bsh[1][arow * 32 + achk] };
    _Float16* bSlot1[2] = { &Bsh[0][(arow + 64) * 32 + achk], &Bsh[1][(arow + 64) * 32 + achk] };

    constexpr bool ASYNC = (USE_ASYNC != 0) && (TRANSB != 0);

    // ---- prologue: stage k=0 into buffer 0 ----
    if (ASYNC) {
        async_cp16(Ag,  aSlot[0]);
        async_cp16(Bg0, bSlot0[0]);
        async_cp16(Bg1, bSlot1[0]);
        async_wait0();
    } else {
        *(uint4*)aSlot[0] = *(const uint4*)Ag;
        if (TRANSB) {
            *(uint4*)bSlot0[0] = *(const uint4*)Bg0;
            *(uint4*)bSlot1[0] = *(const uint4*)Bg1;
        } else {
            #pragma unroll
            for (int i = 0; i < 8; ++i) {
                const int p  = (i << 8) + t;        // 2048 (n,kpair) pairs
                const int nn = p >> 4, kp = (p & 15) << 1;
                const unsigned lo = *(const unsigned short*)(Bm + (long long)kp * ldb + n0 + nn);
                const unsigned hi = *(const unsigned short*)(Bm + (long long)(kp + 1) * ldb + n0 + nn);
                *(unsigned*)&Bsh[0][nn * 32 + kp] = lo | (hi << 16);
            }
        }
    }
    __syncthreads();

    v8f acc[2][2] = {};
    int buf = 0;
    for (int k0 = 0; k0 < K; k0 += 32) {
        const int nk = k0 + 32;
        const bool more = nk < K;
        const int nb = buf ^ 1;

        uint4 ra, rb0, rb1;
        unsigned rbn[8];
        if (more) {
            if (ASYNC) {
                async_cp16(Ag + nk,  aSlot[nb]);
                async_cp16(Bg0 + nk, bSlot0[nb]);
                async_cp16(Bg1 + nk, bSlot1[nb]);
            } else {
                __builtin_prefetch((const void*)(Ag + nk + 32), 0, 1);
                ra = *(const uint4*)(Ag + nk);
                if (TRANSB) {
                    rb0 = *(const uint4*)(Bg0 + nk);
                    rb1 = *(const uint4*)(Bg1 + nk);
                } else {
                    #pragma unroll
                    for (int i = 0; i < 8; ++i) {
                        const int p  = (i << 8) + t;
                        const int nn = p >> 4, kp = (p & 15) << 1;
                        const unsigned lo = *(const unsigned short*)(Bm + (long long)(nk + kp) * ldb + n0 + nn);
                        const unsigned hi = *(const unsigned short*)(Bm + (long long)(nk + kp + 1) * ldb + n0 + nn);
                        rbn[i] = lo | (hi << 16);
                    }
                }
            }
        }

        // ---- compute current buffer: 4 frag loads (2 ds_load_b128 each), 4 WMMAs
        HV a0, a1, b0, b1;
        {
            const _Float16* ab0 = &Ash[buf][(wm * 32 + 0 * 16 + r16) * 32];
            const _Float16* ab1 = &Ash[buf][(wm * 32 + 1 * 16 + r16) * 32];
            a0.q[0] = *(const uint4*)(ab0 + (khalf << 3));
            a0.q[1] = *(const uint4*)(ab0 + 16 + (khalf << 3));
            a1.q[0] = *(const uint4*)(ab1 + (khalf << 3));
            a1.q[1] = *(const uint4*)(ab1 + 16 + (khalf << 3));
            const _Float16* bb0 = &Bsh[buf][(wn * 32 + 0 * 16 + r16) * 32];
            const _Float16* bb1 = &Bsh[buf][(wn * 32 + 1 * 16 + r16) * 32];
            b0.q[0] = *(const uint4*)(bb0 + (khalf << 3));
            b0.q[1] = *(const uint4*)(bb0 + 16 + (khalf << 3));
            b1.q[0] = *(const uint4*)(bb1 + (khalf << 3));
            b1.q[1] = *(const uint4*)(bb1 + 16 + (khalf << 3));
        }
        acc[0][0] = __builtin_amdgcn_wmma_f32_16x16x32_f16(false, a0.v, false, b0.v, (short)0, acc[0][0], false, false);
        acc[0][1] = __builtin_amdgcn_wmma_f32_16x16x32_f16(false, a0.v, false, b1.v, (short)0, acc[0][1], false, false);
        acc[1][0] = __builtin_amdgcn_wmma_f32_16x16x32_f16(false, a1.v, false, b0.v, (short)0, acc[1][0], false, false);
        acc[1][1] = __builtin_amdgcn_wmma_f32_16x16x32_f16(false, a1.v, false, b1.v, (short)0, acc[1][1], false, false);

        if (more) {
            if (!ASYNC) {
                *(uint4*)aSlot[nb] = ra;
                if (TRANSB) {
                    *(uint4*)bSlot0[nb] = rb0;
                    *(uint4*)bSlot1[nb] = rb1;
                } else {
                    #pragma unroll
                    for (int i = 0; i < 8; ++i) {
                        const int p  = (i << 8) + t;
                        const int nn = p >> 4, kp = (p & 15) << 1;
                        *(unsigned*)&Bsh[nb][nn * 32 + kp] = rbn[i];
                    }
                }
            } else {
                async_wait0();
            }
            __syncthreads();
            buf = nb;
        }
    }

    // ---- epilogue: store 2x2 tiles of C ----
    #pragma unroll
    for (int mi = 0; mi < 2; ++mi)
        #pragma unroll
        for (int ni = 0; ni < 2; ++ni)
            #pragma unroll
            for (int r = 0; r < 8; ++r) {
                const int rr = m0 + wm * 32 + mi * 16 + (khalf << 3) + r;
                const int cc = n0 + wn * 32 + ni * 16 + r16;
                C[(long long)rr * ldc + cc] = acc[mi][ni][r];
            }
}

// ---------------------------------------------------------------------
__global__ void k_cvt_f16(const float* __restrict__ src, _Float16* __restrict__ dst, long long n)
{
    long long i = (long long)blockIdx.x * blockDim.x + threadIdx.x;
    if (i < n) dst[i] = (_Float16)src[i];
}

// hidden_states (L,B,S,D) -> ctx_cat16 (B,S, L*D)
__global__ void k_pack_cat(const float* __restrict__ hs, _Float16* __restrict__ cat, long long n)
{
    long long i = (long long)blockIdx.x * blockDim.x + threadIdx.x;
    if (i >= n) return;
    const long long od = i % (LL * DD);
    const long long bs = i / (LL * DD);
    const int l = (int)(od / DD);
    const int d = (int)(od % DD);
    const int b = (int)(bs / SS);
    const int s = (int)(bs % SS);
    cat[i] = (_Float16)hs[(((long long)l * BB + b) * SS + s) * DD + d];
}

__global__ void k_build(const int* __restrict__ ids, const int* __restrict__ anchors,
                        int* __restrict__ pos_kv, int* __restrict__ draft_pos,
                        int* __restrict__ draft_ids, int* __restrict__ labels)
{
    int i = blockIdx.x * blockDim.x + threadIdx.x;
    if (i >= BB * KVV) return;
    const int b = i / KVV, s = i % KVV;
    if (s < SS) { pos_kv[i] = s; return; }
    const int q   = s - SS;
    const int off = q & (BLK - 1);
    const int a   = anchors[b * NAA + (q >> 4)];
    pos_kv[i]              = a + off;
    draft_pos[b * QQ + q]  = a + off;
    draft_ids[b * QQ + q]  = (off == 0) ? ids[b * SS + a] : MASKID;
    labels[b * QQ + q]     = (off == 0) ? -100 : ids[b * SS + a + off];
}

__global__ void k_gather_emb(const float* __restrict__ table, const int* __restrict__ draft_ids,
                             float* __restrict__ emb, _Float16* __restrict__ xkv16, long long n)
{
    long long i = (long long)blockIdx.x * blockDim.x + threadIdx.x;
    if (i >= n) return;
    const int d  = (int)(i % DD);
    const long long bq = i / DD;
    const int b  = (int)(bq / QQ);
    const int q  = (int)(bq % QQ);
    const float v = table[(long long)draft_ids[bq] * DD + d];
    emb[i] = v;
    xkv16[((long long)b * KVV + SS + q) * DD + d] = (_Float16)v;
}

__global__ void k_ctx_to_xkv(const float* __restrict__ ctx, _Float16* __restrict__ xkv16, long long n)
{
    long long i = (long long)blockIdx.x * blockDim.x + threadIdx.x;
    if (i >= n) return;
    const int b = (int)(i / ((long long)SS * DD));
    const long long rem = i % ((long long)SS * DD);
    xkv16[(long long)b * KVV * DD + rem] = (_Float16)ctx[i];
}

// RoPE over f32 linear output, write f16.  n = B*T*H*(DH/2)
__global__ void k_rope(const float* __restrict__ lin, const int* __restrict__ posArr,
                       _Float16* __restrict__ out16, long long n)
{
    long long i = (long long)blockIdx.x * blockDim.x + threadIdx.x;
    if (i >= n) return;
    const int half = DHH / 2;
    const int hd = (int)(i % half);
    long long r  = i / half;
    const int h  = (int)(r % HH);
    const long long bt = r / HH;
    const float inv = expf(-((float)hd / (float)half) * 9.210340371976184f);
    const float ang = (float)posArr[bt] * inv;
    const float c = cosf(ang), sn = sinf(ang);
    const long long base = bt * DD + (long long)h * DHH;
    const float x1 = lin[base + hd];
    const float x2 = lin[base + half + hd];
    out16[base + hd]        = (_Float16)(x1 * c - x2 * sn);
    out16[base + half + hd] = (_Float16)(x2 * c + x1 * sn);
}

// masked softmax over one head's scores [Q, KV]; writes f16 attn
__global__ __launch_bounds__(256)
void k_softmax(const float* __restrict__ sc, _Float16* __restrict__ at,
               const int* __restrict__ anchors, int b)
{
    __shared__ float sm[256];
    const int q = blockIdx.x, t = threadIdx.x;
    const int qb = q >> 4;
    const int apos = anchors[b * NAA + qb];
    const float scale = 0.08838834764831845f;  // 1/sqrt(128)
    const float* row = sc + (long long)q * KVV;

    float mx = -3.0e38f;
    for (int k = t; k < KVV; k += 256) {
        const bool vis = (k < SS) ? (k < apos) : (((k - SS) >> 4) == qb);
        const float v = vis ? row[k] * scale : -1e30f;
        mx = fmaxf(mx, v);
    }
    sm[t] = mx; __syncthreads();
    for (int o = 128; o > 0; o >>= 1) { if (t < o) sm[t] = fmaxf(sm[t], sm[t + o]); __syncthreads(); }
    mx = sm[0]; __syncthreads();

    float sum = 0.f;
    for (int k = t; k < KVV; k += 256) {
        const bool vis = (k < SS) ? (k < apos) : (((k - SS) >> 4) == qb);
        const float v = vis ? row[k] * scale : -1e30f;
        sum += expf(v - mx);
    }
    sm[t] = sum; __syncthreads();
    for (int o = 128; o > 0; o >>= 1) { if (t < o) sm[t] += sm[t + o]; __syncthreads(); }
    const float inv = 1.0f / sm[0];

    for (int k = t; k < KVV; k += 256) {
        const bool vis = (k < SS) ? (k < apos) : (((k - SS) >> 4) == qb);
        const float v = vis ? row[k] * scale : -1e30f;
        at[(long long)q * KVV + k] = (_Float16)(expf(v - mx) * inv);
    }
}

// hidden = emb + attn_out; RMSNorm; write f16
__global__ __launch_bounds__(256)
void k_rmsnorm(const float* __restrict__ emb, const float* __restrict__ ao,
               const float* __restrict__ w, _Float16* __restrict__ h16)
{
    __shared__ float sm[256];
    const long long r = blockIdx.x;
    const int t = threadIdx.x;
    const float* e = emb + r * DD;
    const float* a = ao  + r * DD;
    float ss = 0.f;
    for (int d = t; d < DD; d += 256) { const float v = e[d] + a[d]; ss += v * v; }
    sm[t] = ss; __syncthreads();
    for (int o = 128; o > 0; o >>= 1) { if (t < o) sm[t] += sm[t + o]; __syncthreads(); }
    const float scale = rsqrtf(sm[0] / (float)DD + EPS_);
    for (int d = t; d < DD; d += 256)
        h16[r * DD + d] = (_Float16)((e[d] + a[d]) * scale * w[d]);
}

__global__ void k_init_state(float* rmax, float* rsum, float* bval, int* bidx, float* llog)
{
    int i = blockIdx.x * blockDim.x + threadIdx.x;
    if (i < BB * QQ) { rmax[i] = -3e38f; rsum[i] = 0.f; bval[i] = -3e38f; bidx[i] = 0; llog[i] = 0.f; }
}

// online log-softmax / argmax update from one logits chunk [rows, cn]
__global__ __launch_bounds__(256)
void k_chunk_update(const float* __restrict__ chunk, int n0, int cn,
                    const int* __restrict__ labels,
                    float* __restrict__ rmax, float* __restrict__ rsum,
                    float* __restrict__ bval, int* __restrict__ bidx,
                    float* __restrict__ llog)
{
    __shared__ float smx[256], ssm[256], sbv[256];
    __shared__ int   sbi[256];
    const int r = blockIdx.x, t = threadIdx.x;
    const float* row = chunk + (long long)r * cn;

    float m = -3e38f, s = 0.f, bv = -3e38f; int bi = n0;
    for (int c = t; c < cn; c += 256) {
        const float v = row[c];
        if (v > m) { s = s * expf(m - v) + 1.0f; m = v; }
        else       { s += expf(v - m); }
        const int gi = n0 + c;
        if (v > bv || (v == bv && gi < bi)) { bv = v; bi = gi; }
    }
    smx[t] = m; ssm[t] = s; sbv[t] = bv; sbi[t] = bi;
    __syncthreads();
    for (int o = 128; o > 0; o >>= 1) {
        if (t < o) {
            const float m1 = smx[t], m2 = smx[t + o];
            const float mm = fmaxf(m1, m2);
            ssm[t] = ssm[t] * expf(m1 - mm) + ssm[t + o] * expf(m2 - mm);
            smx[t] = mm;
            const float v2 = sbv[t + o]; const int i2 = sbi[t + o];
            if (v2 > sbv[t] || (v2 == sbv[t] && i2 < sbi[t])) { sbv[t] = v2; sbi[t] = i2; }
        }
        __syncthreads();
    }
    if (t == 0) {
        const int lab = labels[r];
        const int l = (lab < 0) ? 0 : lab;
        if (l >= n0 && l < n0 + cn) llog[r] = row[l - n0];
        const float m1 = rmax[r], m2 = smx[0];
        const float mm = fmaxf(m1, m2);
        rsum[r] = rsum[r] * expf(m1 - mm) + ssm[0] * expf(m2 - mm);
        rmax[r] = mm;
        const float v2 = sbv[0]; const int i2 = sbi[0];
        if (v2 > bval[r] || (v2 == bval[r] && i2 < bidx[r])) { bval[r] = v2; bidx[r] = i2; }
    }
}

__global__ __launch_bounds__(256)
void k_final(const float* __restrict__ rmax, const float* __restrict__ rsum,
             const float* __restrict__ llog, const int* __restrict__ bidx,
             const int* __restrict__ labels, float* __restrict__ out)
{
    __shared__ float s0[256], s1[256], s2[256], s3[256];
    const int t = threadIdx.x;
    float lnum = 0.f, wsum = 0.f, anum = 0.f, vcnt = 0.f;
    for (int r = t; r < BB * QQ; r += 256) {
        const int lab = labels[r];
        const bool valid = (lab != -100);
        const float nll = -(llog[r] - rmax[r] - logf(rsum[r]));
        const int off = r & (BLK - 1);
        const float decay = (off == 0) ? 0.0f : expf(-((float)off - 1.0f) / GAMMA_);
        const float w = valid ? decay : 0.0f;
        lnum += nll * w;
        wsum += w;
        if (valid) { vcnt += 1.0f; if (bidx[r] == lab) anum += 1.0f; }
    }
    s0[t] = lnum; s1[t] = wsum; s2[t] = anum; s3[t] = vcnt; __syncthreads();
    for (int o = 128; o > 0; o >>= 1) {
        if (t < o) { s0[t] += s0[t+o]; s1[t] += s1[t+o]; s2[t] += s2[t+o]; s3[t] += s3[t+o]; }
        __syncthreads();
    }
    if (t == 0) {
        out[0] = s0[0] / fmaxf(s1[0], 1e-6f);
        out[1] = s2[0] / fmaxf(s3[0], 1.0f);
    }
}

// =====================================================================
extern "C" void kernel_launch(void* const* d_in, const int* in_sizes, int n_in,
                              void* d_out, int out_size, void* d_ws, size_t ws_size,
                              hipStream_t stream)
{
    const int*   input_ids = (const int*)  d_in[0];
    const float* hs        = (const float*)d_in[1];
    const float* lm_w      = (const float*)d_in[3];
    const float* norm_w    = (const float*)d_in[4];
    const int*   anchors   = (const int*)  d_in[5];
    const float* ctx_w     = (const float*)d_in[6];
    const float* embed     = (const float*)d_in[7];
    const float* wq        = (const float*)d_in[8];
    const float* wk        = (const float*)d_in[9];
    const float* wv        = (const float*)d_in[10];
    const float* wo        = (const float*)d_in[11];
    float* out = (float*)d_out;

    char* p = (char*)d_ws;
    auto alloc = [&](size_t bytes) -> char* {
        char* r = p; p += (bytes + 255) & ~(size_t)255; return r;
    };
    const long long LD = (long long)LL * DD;

    _Float16* cat16   = (_Float16*)alloc((long long)BB * SS * LD * 2);
    _Float16* ctxw16  = (_Float16*)alloc((long long)DD * LD * 2);
    _Float16* wq16    = (_Float16*)alloc((long long)DD * DD * 2);
    _Float16* wk16    = (_Float16*)alloc((long long)DD * DD * 2);
    _Float16* wv16    = (_Float16*)alloc((long long)DD * DD * 2);
    _Float16* wo16    = (_Float16*)alloc((long long)DD * DD * 2);
    _Float16* lm16    = (_Float16*)alloc((long long)VOC * DD * 2);
    _Float16* xkv16   = (_Float16*)alloc((long long)BB * KVV * DD * 2);
    float*    emb     = (float*)   alloc((long long)BB * QQ * DD * 4);
    float*    ctx     = (float*)   alloc((long long)BB * SS * DD * 4);
    float*    qlin    = (float*)   alloc((long long)BB * QQ * DD * 4);
    float*    klin    = (float*)   alloc((long long)BB * KVV * DD * 4);
    float*    vlin    = (float*)   alloc((long long)BB * KVV * DD * 4);
    _Float16* q16     = (_Float16*)alloc((long long)BB * QQ * DD * 2);
    _Float16* k16     = (_Float16*)alloc((long long)BB * KVV * DD * 2);
    _Float16* v16     = (_Float16*)alloc((long long)BB * KVV * DD * 2);
    float*    scores  = (float*)   alloc((long long)QQ * KVV * 4);
    _Float16* attn16  = (_Float16*)alloc((long long)QQ * KVV * 2);
    float*    outlin  = (float*)   alloc((long long)BB * QQ * DD * 4);
    _Float16* out16   = (_Float16*)alloc((long long)BB * QQ * DD * 2);
    float*    attnout = (float*)   alloc((long long)BB * QQ * DD * 4);
    _Float16* hid16   = (_Float16*)alloc((long long)BB * QQ * DD * 2);
    float*    chunk   = (float*)   alloc((long long)BB * QQ * CHUNKW * 4);
    float*    rmax    = (float*)   alloc((long long)BB * QQ * 4);
    float*    rsum    = (float*)   alloc((long long)BB * QQ * 4);
    float*    bvalv   = (float*)   alloc((long long)BB * QQ * 4);
    int*      bidxv   = (int*)     alloc((long long)BB * QQ * 4);
    float*    llogv   = (float*)   alloc((long long)BB * QQ * 4);
    int*      pos_kv  = (int*)     alloc((long long)BB * KVV * 4);
    int*      dpos    = (int*)     alloc((long long)BB * QQ * 4);
    int*      dids    = (int*)     alloc((long long)BB * QQ * 4);
    int*      labels  = (int*)     alloc((long long)BB * QQ * 4);

    auto ew = [&](long long n) { return dim3((unsigned)((n + 255) / 256)); };

    auto gemmT = [&](const _Float16* A, long long lda, long long sAz,
                     const _Float16* B, long long ldb, long long sBz,
                     float* C, long long ldc, long long sCz,
                     int M, int N, int K, int Z) {
        dim3 g((unsigned)((M / 64) * (N / 128)), 1, (unsigned)Z);
        gemm_wmma_blk<1><<<g, 256, 0, stream>>>(A, lda, sAz, B, ldb, sBz, C, ldc, sCz, M, N, K);
    };
    auto gemmN = [&](const _Float16* A, long long lda, long long sAz,
                     const _Float16* B, long long ldb, long long sBz,
                     float* C, long long ldc, long long sCz,
                     int M, int N, int K, int Z) {
        dim3 g((unsigned)((M / 64) * (N / 128)), 1, (unsigned)Z);
        gemm_wmma_blk<0><<<g, 256, 0, stream>>>(A, lda, sAz, B, ldb, sBz, C, ldc, sCz, M, N, K);
    };

    // 1) convert weights to f16
    k_cvt_f16<<<ew((long long)DD * LD), 256, 0, stream>>>(ctx_w, ctxw16, (long long)DD * LD);
    k_cvt_f16<<<ew((long long)DD * DD), 256, 0, stream>>>(wq, wq16, (long long)DD * DD);
    k_cvt_f16<<<ew((long long)DD * DD), 256, 0, stream>>>(wk, wk16, (long long)DD * DD);
    k_cvt_f16<<<ew((long long)DD * DD), 256, 0, stream>>>(wv, wv16, (long long)DD * DD);
    k_cvt_f16<<<ew((long long)DD * DD), 256, 0, stream>>>(wo, wo16, (long long)DD * DD);
    k_cvt_f16<<<ew((long long)VOC * DD), 256, 0, stream>>>(lm_w, lm16, (long long)VOC * DD);

    // 2) ctx_cat pack (L,B,S,D -> B,S,L*D) as f16
    k_pack_cat<<<ew((long long)BB * SS * LD), 256, 0, stream>>>(hs, cat16, (long long)BB * SS * LD);

    // 3) draft ids / positions / labels
    k_build<<<ew(BB * KVV), 256, 0, stream>>>(input_ids, anchors, pos_kv, dpos, dids, labels);

    // 4) embedding gather
    k_gather_emb<<<ew((long long)BB * QQ * DD), 256, 0, stream>>>(
        embed, dids, emb, xkv16, (long long)BB * QQ * DD);

    // 5) ctx = ctx_cat @ ctx_proj_w^T
    gemmT(cat16, LD, 0, ctxw16, LD, 0, ctx, DD, 0, BB * SS, DD, (int)LD, 1);
    k_ctx_to_xkv<<<ew((long long)BB * SS * DD), 256, 0, stream>>>(ctx, xkv16, (long long)BB * SS * DD);

    // 6) q/k/v projections (batched over B via z-stride)
    gemmT(xkv16 + (long long)SS * DD, DD, (long long)KVV * DD, wq16, DD, 0,
          qlin, DD, (long long)QQ * DD, QQ, DD, DD, BB);
    gemmT(xkv16, DD, (long long)KVV * DD, wk16, DD, 0,
          klin, DD, (long long)KVV * DD, KVV, DD, DD, BB);
    gemmT(xkv16, DD, (long long)KVV * DD, wv16, DD, 0,
          vlin, DD, (long long)KVV * DD, KVV, DD, DD, BB);

    // 7) RoPE + v convert
    k_rope<<<ew((long long)BB * QQ * HH * (DHH / 2)), 256, 0, stream>>>(
        qlin, dpos, q16, (long long)BB * QQ * HH * (DHH / 2));
    k_rope<<<ew((long long)BB * KVV * HH * (DHH / 2)), 256, 0, stream>>>(
        klin, pos_kv, k16, (long long)BB * KVV * HH * (DHH / 2));
    k_cvt_f16<<<ew((long long)BB * KVV * DD), 256, 0, stream>>>(vlin, v16, (long long)BB * KVV * DD);

    // 8) attention per (b, h)
    for (int b = 0; b < BB; ++b) {
        for (int h = 0; h < HH; ++h) {
            const long long qoff = (long long)b * QQ * DD + (long long)h * DHH;
            const long long koff = (long long)b * KVV * DD + (long long)h * DHH;
            gemmT(q16 + qoff, DD, 0, k16 + koff, DD, 0, scores, KVV, 0, QQ, KVV, DHH, 1);
            k_softmax<<<QQ, 256, 0, stream>>>(scores, attn16, anchors, b);
            gemmN(attn16, KVV, 0, v16 + koff, DD, 0, outlin + qoff, DD, 0, QQ, DHH, KVV, 1);
        }
    }

    // 9) output projection
    k_cvt_f16<<<ew((long long)BB * QQ * DD), 256, 0, stream>>>(outlin, out16, (long long)BB * QQ * DD);
    gemmT(out16, DD, 0, wo16, DD, 0, attnout, DD, 0, BB * QQ, DD, DD, 1);

    // 10) residual + RMSNorm
    k_rmsnorm<<<BB * QQ, 256, 0, stream>>>(emb, attnout, norm_w, hid16);

    // 11) lm_head streamed: 15 chunks of 2048 cols + final 1280
    k_init_state<<<ew(BB * QQ), 256, 0, stream>>>(rmax, rsum, bvalv, bidxv, llogv);
    for (int c = 0; c < NCHUNK; ++c) {
        const int n0 = c * CHUNKW;
        const int cn = (c == NCHUNK - 1) ? (VOC - n0) : CHUNKW;   // 1280 on last
        gemmT(hid16, DD, 0, lm16 + (long long)n0 * DD, DD, 0, chunk, cn, 0, BB * QQ, cn, DD, 1);
        k_chunk_update<<<BB * QQ, 256, 0, stream>>>(chunk, n0, cn, labels, rmax, rsum, bvalv, bidxv, llogv);
    }

    // 12) final loss + accuracy
    k_final<<<1, 256, 0, stream>>>(rmax, rsum, llogv, bidxv, labels, out);
}
